// GWLoss_10720238370848
// MI455X (gfx1250) — matrix operand order repaired and checked by reading, not compiled, not run on hardware
//
#include <hip/hip_runtime.h>
#include <stdint.h>

// ---------------- types ----------------
typedef __attribute__((ext_vector_type(16))) __bf16 v16bf;
typedef __attribute__((ext_vector_type(8)))  __bf16 v8bf;
typedef __attribute__((ext_vector_type(8)))  float  v8f;
typedef int vi4 __attribute__((vector_size(16)));

union AFrag { v16bf v; v8bf h[2]; };

__device__ __forceinline__ float bf2f(__bf16 x) {
    unsigned short u = __builtin_bit_cast(unsigned short, x);
    unsigned int w = ((unsigned int)u) << 16;
    return __builtin_bit_cast(float, w);
}
__device__ __forceinline__ __bf16 f2bf(float f) {
    unsigned int w = __builtin_bit_cast(unsigned int, f);
    unsigned int r = (w + 0x7FFFu + ((w >> 16) & 1u)) >> 16;
    return __builtin_bit_cast(__bf16, (unsigned short)r);
}

#ifndef __has_builtin
#define __has_builtin(x) 0
#endif

#if __has_builtin(__builtin_amdgcn_global_load_async_to_lds_b128)
#define ASYNC_LDS 1
#else
#define ASYNC_LDS 0
#endif

#if ASYNC_LDS
typedef __attribute__((address_space(1))) vi4* as1_vi4p;
typedef __attribute__((address_space(3))) vi4* as3_vi4p;
__device__ __forceinline__ void async_b128(const __bf16* g, __bf16* s) {
    __builtin_amdgcn_global_load_async_to_lds_b128(
        (as1_vi4p)(vi4*)(void*)g,
        (as3_vi4p)(vi4*)(void*)s, 0, 0);
}
__device__ __forceinline__ void wait_async0() {
#if __has_builtin(__builtin_amdgcn_s_wait_asynccnt)
    __builtin_amdgcn_s_wait_asynccnt(0);
#else
    asm volatile("s_wait_asynccnt 0x0" ::: "memory");
#endif
}
#endif

#define NMAT 4096
#define DFEAT 256

enum { MODE_CT = 0, MODE_CS2 = 1, MODE_G1 = 2, MODE_K = 3, MODE_LOSS = 4 };

// ---------------- row normalize: Y[i,:] = (X[i,:]-mean)/||.|| as bf16 ----------------
__global__ void gw_normalize(const float* __restrict__ X, __bf16* __restrict__ Y) {
    __shared__ float sm[256];
    const int t = threadIdx.x;
    const size_t base = (size_t)blockIdx.x * DFEAT;
    float x = X[base + t];
    sm[t] = x; __syncthreads();
    for (int s = 128; s; s >>= 1) { if (t < s) sm[t] += sm[t + s]; __syncthreads(); }
    const float mean = sm[0] * (1.0f / DFEAT);
    __syncthreads();
    const float xc = x - mean;
    sm[t] = xc * xc; __syncthreads();
    for (int s = 128; s; s >>= 1) { if (t < s) sm[t] += sm[t + s]; __syncthreads(); }
    const float inv = rsqrtf(sm[0]);
    Y[base + t] = f2bf(xc * inv);
}

// ---------------- WMMA GEMM: D[m][n] = sum_k A[m][k] * BT[n][k], fused epilogues ----------------
__global__ __launch_bounds__(256)
void gw_gemm(const __bf16* __restrict__ A, const __bf16* __restrict__ BT, int K, int mode,
             __bf16* __restrict__ outB, float* __restrict__ KfOut,
             const float* __restrict__ av, const float* __restrict__ bv,
             const float* __restrict__ uu, const float* __restrict__ vv,
             const float* __restrict__ KfIn, float* __restrict__ loss) {
    constexpr int APAD = 40;           // 32 + 8 pad (80B rows, bank-conflict-free b128 frags)
    __shared__ __bf16 Asm[2][128 * APAD];
    __shared__ __bf16 Bsm[2][128 * APAD];
    __shared__ float red[256];

    const int t   = threadIdx.x;
    const int bm0 = blockIdx.y << 7;
    const int bn0 = blockIdx.x << 7;

    // cooperative tile loaders: thread -> (row, 16-elem half)
    const int lr = t >> 1;
    const int lc = (t & 1) << 4;
    const __bf16* gA = A  + (size_t)(bm0 + lr) * K + lc;
    const __bf16* gB = BT + (size_t)(bn0 + lr) * K + lc;

    const int lane = t & 31;
    const int wid  = t >> 5;
    const int wm   = (wid >> 2) << 6;  // 0 / 64
    const int wn   = (wid & 3) << 5;   // 0..96
    const int half = lane >> 4;
    const int l16  = lane & 15;

    v8f acc[4][2];
#pragma unroll
    for (int i = 0; i < 4; ++i)
#pragma unroll
        for (int j = 0; j < 2; ++j)
            acc[i][j] = (v8f){0.f, 0.f, 0.f, 0.f, 0.f, 0.f, 0.f, 0.f};

    const int nk = K >> 5;
    { // prologue: k-tile 0 -> buffer 0
        __bf16* sa = &Asm[0][lr * APAD + lc];
        __bf16* sb = &Bsm[0][lr * APAD + lc];
#if ASYNC_LDS
        async_b128(gA, sa);       async_b128(gA + 8, sa + 8);
        async_b128(gB, sb);       async_b128(gB + 8, sb + 8);
        wait_async0();
#else
        uint4 a0 = *(const uint4*)(gA);     uint4 a1 = *(const uint4*)(gA + 8);
        uint4 b0 = *(const uint4*)(gB);     uint4 b1 = *(const uint4*)(gB + 8);
        *(uint4*)sa = a0; *(uint4*)(sa + 8) = a1;
        *(uint4*)sb = b0; *(uint4*)(sb + 8) = b1;
#endif
    }
    __syncthreads();

    for (int kt = 0; kt < nk; ++kt) {
        const int cur = kt & 1;
        const bool pf = (kt + 1) < nk;
#if ASYNC_LDS
        if (pf) { // direct global -> LDS async copy into the other buffer
            const __bf16* pa = gA + ((kt + 1) << 5);
            const __bf16* pb = gB + ((kt + 1) << 5);
            __bf16* da = &Asm[cur ^ 1][lr * APAD + lc];
            __bf16* db = &Bsm[cur ^ 1][lr * APAD + lc];
            async_b128(pa, da);   async_b128(pa + 8, da + 8);
            async_b128(pb, db);   async_b128(pb + 8, db + 8);
        }
#else
        uint4 a0, a1, b0, b1;
        if (pf) {
            const __bf16* pa = gA + ((kt + 1) << 5);
            const __bf16* pb = gB + ((kt + 1) << 5);
            a0 = *(const uint4*)pa; a1 = *(const uint4*)(pa + 8);
            b0 = *(const uint4*)pb; b1 = *(const uint4*)(pb + 8);
        }
#endif
        // per-lane fragments from LDS (layouts per CDNA5 ISA 7.12.2)
        const __bf16* sa = &Asm[cur][(wm + l16) * APAD + (half << 3)];
        const __bf16* sb = &Bsm[cur][(wn + l16) * APAD + (half << 3)];
        AFrag af[4], bfg[2];
#pragma unroll
        for (int tm = 0; tm < 4; ++tm) {
            af[tm].h[0] = *(const v8bf*)(sa + tm * 16 * APAD);        // K = 8h..8h+7
            af[tm].h[1] = *(const v8bf*)(sa + tm * 16 * APAD + 16);   // K = 16+8h..
        }
#pragma unroll
        for (int tn = 0; tn < 2; ++tn) {
            bfg[tn].h[0] = *(const v8bf*)(sb + tn * 16 * APAD);       // K = 8h..8h+7
            bfg[tn].h[1] = *(const v8bf*)(sb + tn * 16 * APAD + 16);  // K = 16+8h..
        }
#pragma unroll
        for (int tm = 0; tm < 4; ++tm)
#pragma unroll
            for (int tn = 0; tn < 2; ++tn)
                acc[tm][tn] = __builtin_amdgcn_wmma_f32_16x16x32_bf16(
                    false, af[tm].v, false, bfg[tn].v, (short)0, acc[tm][tn], false, false);
#if ASYNC_LDS
        wait_async0();           // next-buffer tiles landed in LDS
        __syncthreads();
#else
        if (pf) {
            __bf16* da = &Asm[cur ^ 1][lr * APAD + lc];
            __bf16* db = &Bsm[cur ^ 1][lr * APAD + lc];
            *(uint4*)da = a0; *(uint4*)(da + 8) = a1;
            *(uint4*)db = b0; *(uint4*)(db + 8) = b1;
        }
        __syncthreads();
#endif
    }

    // epilogue
    float lsum = 0.f;
#pragma unroll
    for (int tm = 0; tm < 4; ++tm) {
#pragma unroll
        for (int tn = 0; tn < 2; ++tn) {
            const int n = bn0 + wn + (tn << 4) + l16;
#pragma unroll
            for (int r = 0; r < 8; ++r) {
                const int m = bm0 + wm + (tm << 4) + (half << 3) + r;
                const float a = acc[tm][tn][r];
                const size_t idx = ((size_t)m << 12) + n;
                if (mode == MODE_CT)        outB[idx] = f2bf(1.0f - a);
                else if (mode == MODE_CS2)  outB[idx] = f2bf(2.0f * (1.0f - a));
                else if (mode == MODE_G1)   outB[idx] = f2bf(a);
                else if (mode == MODE_K)    KfOut[idx] = expf(-2.0f * (av[m] + bv[n] - a));
                else { // MODE_LOSS: sum(tens * T), T = u K v
                    const float tens = av[m] + bv[n] - a;
                    lsum += tens * (uu[m] * KfIn[idx] * vv[n]);
                }
            }
        }
    }
    if (mode == MODE_LOSS) {
        red[t] = lsum; __syncthreads();
        for (int s = 128; s; s >>= 1) { if (t < s) red[t] += red[t + s]; __syncthreads(); }
        if (t == 0) atomicAdd(loss, red[0]);
    }
}

// ---------------- a[m] = (1/N) * sum_k (scale*Mb[m][k])^2  (wave per row) ----------------
__global__ void gw_rowsq(const __bf16* __restrict__ Mb, float* __restrict__ out, float scale) {
    const int row  = (blockIdx.x << 3) + (threadIdx.x >> 5);
    const int lane = threadIdx.x & 31;
    const __bf16* p = Mb + ((size_t)row << 12);
    float s = 0.f;
#pragma unroll 4
    for (int it = 0; it < 16; ++it) {
        const v8bf c = *(const v8bf*)(p + it * 256 + lane * 8);
#pragma unroll
        for (int e = 0; e < 8; ++e) { const float x = scale * bf2f(c[e]); s += x * x; }
    }
    for (int o = 16; o; o >>= 1) s += __shfl_xor(s, o);
    if (lane == 0) out[row] = s * (1.0f / NMAT);
}

// ---------------- TT[j][i] = bf16(u[i]*K[i][j]*v[j])  (fused T-build + transpose) ----------------
__global__ void gw_ttrans(const float* __restrict__ Kf, const float* __restrict__ u,
                          const float* __restrict__ v, __bf16* __restrict__ TT) {
    __shared__ float sm[32][33];
    const int tx = threadIdx.x & 31, ty = threadIdx.x >> 5;
    const int i0 = blockIdx.y << 5, j0 = blockIdx.x << 5;
    for (int yy = ty; yy < 32; yy += 8) {
        const int i = i0 + yy, j = j0 + tx;
        sm[yy][tx] = u[i] * Kf[((size_t)i << 12) + j] * v[j];
    }
    __syncthreads();
    for (int yy = ty; yy < 32; yy += 8)
        TT[((size_t)(j0 + yy) << 12) + i0 + tx] = f2bf(sm[tx][yy]);
}

// ---------------- sinkhorn pieces ----------------
// wpart[chunk][j] = sum_{i in chunk of 256} K[i][j]*u[i]
__global__ void gw_colpart(const float* __restrict__ Kf, const float* __restrict__ u,
                           float* __restrict__ wpart) {
    const int j  = (blockIdx.x << 8) + threadIdx.x;
    const int i0 = blockIdx.y << 8;
    float s = 0.f;
#pragma unroll 4
    for (int i = i0; i < i0 + 256; ++i) s += Kf[((size_t)i << 12) + j] * u[i];
    wpart[((size_t)blockIdx.y << 12) + j] = s;
}
// v[j] = q / sum_c wpart[c][j]
__global__ void gw_vdiv(const float* __restrict__ wpart, float* __restrict__ v) {
    const int j = (blockIdx.x << 8) + threadIdx.x;
    float s = 0.f;
#pragma unroll
    for (int c = 0; c < 16; ++c) s += wpart[((size_t)c << 12) + j];
    v[j] = (1.0f / NMAT) / s;
}
// u[i] = p / sum_j K[i][j]*v[j]   (wave per row, vectorized)
__global__ void gw_rowsum_u(const float* __restrict__ Kf, const float* __restrict__ v,
                            float* __restrict__ u) {
    const int row  = (blockIdx.x << 3) + (threadIdx.x >> 5);
    const int lane = threadIdx.x & 31;
    const float* p = Kf + ((size_t)row << 12);
    float s = 0.f;
#pragma unroll 4
    for (int it = 0; it < 32; ++it) {
        const int j = it * 128 + lane * 4;
        const float4 k4 = *(const float4*)(p + j);
        const float4 v4 = *(const float4*)(v + j);
        s += k4.x * v4.x + k4.y * v4.y + k4.z * v4.z + k4.w * v4.w;
    }
    for (int o = 16; o; o >>= 1) s += __shfl_xor(s, o);
    if (lane == 0) u[row] = (1.0f / NMAT) / s;
}

// ---------------- fills and final T output ----------------
__global__ void gw_fill_bf(__bf16* __restrict__ p, float val) {
    p[((size_t)blockIdx.x << 8) + threadIdx.x] = f2bf(val);
}
__global__ void gw_fill_f32(float* __restrict__ p, float val) {
    p[((size_t)blockIdx.x << 8) + threadIdx.x] = val;
}
__global__ void gw_write_T(const float* __restrict__ Kf, const float* __restrict__ u,
                           const float* __restrict__ v, float* __restrict__ out) {
    const size_t id = ((size_t)blockIdx.x << 8) + threadIdx.x;
    const int i = (int)(id >> 12), j = (int)(id & 4095);
    out[1 + id] = u[i] * Kf[id] * v[j];
    if (id == 0) out[0] = 0.0f;   // loss accumulator, filled by MODE_LOSS gemm after this
}

// ---------------- host orchestration ----------------
extern "C" void kernel_launch(void* const* d_in, const int* in_sizes, int n_in,
                              void* d_out, int out_size, void* d_ws, size_t ws_size,
                              hipStream_t stream) {
    (void)in_sizes; (void)n_in; (void)out_size; (void)ws_size;
    const float* ft = (const float*)d_in[0];
    const float* fs = (const float*)d_in[1];
    float* out = (float*)d_out;

    uint8_t* w = (uint8_t*)d_ws;
    size_t off = 0;
    auto carve = [&](size_t bytes) -> void* {
        void* p = w + off;
        off = (off + bytes + 255) & ~(size_t)255;
        return p;
    };
    const size_t NN = (size_t)NMAT * NMAT;
    __bf16* Yt   = (__bf16*)carve((size_t)NMAT * DFEAT * 2);
    __bf16* Ys   = (__bf16*)carve((size_t)NMAT * DFEAT * 2);
    __bf16* CtB  = (__bf16*)carve(NN * 2);
    __bf16* Cs2B = (__bf16*)carve(NN * 2);
    __bf16* TTb  = (__bf16*)carve(NN * 2);
    __bf16* G1b  = (__bf16*)carve(NN * 2);
    float*  Kf   = (float*)carve(NN * 4);
    float*  av   = (float*)carve(NMAT * 4);
    float*  bvv  = (float*)carve(NMAT * 4);
    float*  uvec = (float*)carve(NMAT * 4);
    float*  vvec = (float*)carve(NMAT * 4);
    float*  wpart= (float*)carve((size_t)16 * NMAT * 4);

    const dim3 g128(32, 32);

    // correlation costs: Ct = 1 - corr(ft), 2Cs = 2(1 - corr(fs))
    gw_normalize<<<NMAT, 256, 0, stream>>>(ft, Yt);
    gw_normalize<<<NMAT, 256, 0, stream>>>(fs, Ys);
    gw_gemm<<<g128, 256, 0, stream>>>(Yt, Yt, DFEAT, MODE_CT, CtB, nullptr,
                                      nullptr, nullptr, nullptr, nullptr, nullptr, nullptr);
    gw_gemm<<<g128, 256, 0, stream>>>(Ys, Ys, DFEAT, MODE_CS2, Cs2B, nullptr,
                                      nullptr, nullptr, nullptr, nullptr, nullptr, nullptr);
    gw_rowsq<<<512, 256, 0, stream>>>(CtB, av, 1.0f);
    gw_rowsq<<<512, 256, 0, stream>>>(Cs2B, bvv, 0.5f);

    // T0 = p q^T = 1/n^2 (only needed transposed, in bf16)
    gw_fill_bf<<<65536, 256, 0, stream>>>(TTb, 1.0f / ((float)NMAT * (float)NMAT));

    for (int it = 0; it < 10; ++it) {
        if (it > 0)
            gw_ttrans<<<dim3(128, 128), 256, 0, stream>>>(Kf, uvec, vvec, TTb);
        // G1 = Ct @ T
        gw_gemm<<<g128, 256, 0, stream>>>(CtB, TTb, NMAT, MODE_G1, G1b, nullptr,
                                          nullptr, nullptr, nullptr, nullptr, nullptr, nullptr);
        // K = exp(-2*(constC - G1 @ 2Cs))
        gw_gemm<<<g128, 256, 0, stream>>>(G1b, Cs2B, NMAT, MODE_K, nullptr, Kf,
                                          av, bvv, nullptr, nullptr, nullptr, nullptr);
        // sinkhorn, u0 = 1
        gw_fill_f32<<<16, 256, 0, stream>>>(uvec, 1.0f);
        for (int s = 0; s < 50; ++s) {
            gw_colpart<<<dim3(16, 16), 256, 0, stream>>>(Kf, uvec, wpart);
            gw_vdiv<<<16, 256, 0, stream>>>(wpart, vvec);
            gw_rowsum_u<<<512, 256, 0, stream>>>(Kf, vvec, uvec);
        }
        gw_colpart<<<dim3(16, 16), 256, 0, stream>>>(Kf, uvec, wpart);
        gw_vdiv<<<16, 256, 0, stream>>>(wpart, vvec);
        // T = diag(u) K diag(v) — kept implicit in (Kf,u,v)
    }

    // final: tens from final T, loss = sum(tens * T), emit T
    gw_ttrans<<<dim3(128, 128), 256, 0, stream>>>(Kf, uvec, vvec, TTb);
    gw_gemm<<<g128, 256, 0, stream>>>(CtB, TTb, NMAT, MODE_G1, G1b, nullptr,
                                      nullptr, nullptr, nullptr, nullptr, nullptr, nullptr);
    gw_write_T<<<65536, 256, 0, stream>>>(Kf, uvec, vvec, out);  // also zeroes out[0]
    gw_gemm<<<g128, 256, 0, stream>>>(G1b, Cs2B, NMAT, MODE_LOSS, nullptr, nullptr,
                                      av, bvv, uvec, vvec, Kf, out);
}
